// LowrankVideo_62371515072639
// MI455X (gfx1250) — compile-verified, matrix operand order checked by reading, not run on hardware
//
#include <hip/hip_runtime.h>
#include <math.h>

typedef __attribute__((ext_vector_type(16))) _Float16 v16h;
typedef __attribute__((ext_vector_type(8)))  float    v8f;
typedef __attribute__((ext_vector_type(4)))  int      v4i_t;

#define RR   128      // spatial grid resolution
#define TTT  150      // time grid resolution
#define FF   32       // feature channels

#if __has_builtin(__builtin_amdgcn_global_load_async_to_lds_b128) && \
    __has_builtin(__builtin_amdgcn_s_wait_asynccnt)
#define HAVE_ASYNC_LDS 1
#else
#define HAVE_ASYNC_LDS 0
#endif

union FragU { v16h h; uint4 q[2]; };

__device__ __forceinline__ v8f wmma16x16x32(v16h a, v16h b, v8f c) {
  return __builtin_amdgcn_wmma_f32_16x16x32_f16(false, a, false, b, (short)0, c,
                                                false, false);
}

// B (pre-fragmented) fragment: 32 contiguous bytes per lane.
__device__ __forceinline__ v16h ld_frag(const _Float16* p) {
  FragU f;
  f.q[0] = *reinterpret_cast<const uint4*>(p);
  f.q[1] = *reinterpret_cast<const uint4*>(p + 8);
  return f.h;
}

// A-matrix 16xK fragment from a row-major [16][ldh] half tile.
// lane<16 holds K=kStart+0..7 and kStart+16..23 for row=lane;
// lane>=16 holds K=kStart+8..15 and kStart+24..31 for row=lane-16.
__device__ __forceinline__ v16h ld_afrag(const _Float16* base, int row, int ldh,
                                         int kStart, int hi) {
  const _Float16* p0 = base + row * ldh + kStart + hi * 8;
  FragU f;
  f.q[0] = *reinterpret_cast<const uint4*>(p0);
  f.q[1] = *reinterpret_cast<const uint4*>(p0 + 16);
  return f.h;
}

// ---------------------------------------------------------------------------
// Kernel 0: build the fragment-ordered f16 weight image (8192 halves = 16KB)
// in workspace, once per launch. Layout (halves):
//   [0..4095]    W1  frags (kt0..1 x nt0..3)
//   [4096..5119] W2  frags (kt0..1)
//   [5120..7167] Wc1 frags (nt0..3)
//   [7168..8191] Wc2 frags (kt0..1)
// ---------------------------------------------------------------------------
__device__ __forceinline__ void fill_frags(_Float16* dst, const float* W,
                                           int Kact, int Nact, int Ntot,
                                           int nkt, int nnt, int tid, int nth) {
  int total = nkt * nnt * 512;
  for (int idx = tid; idx < total; idx += nth) {
    int frag = idx >> 9, within = idx & 511;
    int kt = frag / nnt, nt = frag % nnt;
    int l = within >> 4, i = within & 15;
    int kl = (i < 8 ? i : i + 8) + ((l >= 16) ? 8 : 0);
    int k = kt * 32 + kl;
    int nc = nt * 16 + (l & 15);
    float v = (k < Kact && nc < Nact) ? W[k * Ntot + nc] : 0.f;
    dst[idx] = (_Float16)v;
  }
}

__global__ void k_prepw(const float* __restrict__ W1, const float* __restrict__ W2,
                        const float* __restrict__ Wc1, const float* __restrict__ Wc2,
                        _Float16* __restrict__ Wfrag) {
  int tid = threadIdx.x;
  fill_frags(Wfrag + 0,    W1,  35, 64, 64, 2, 4, tid, 256);
  fill_frags(Wfrag + 4096, W2,  64, 16, 16, 2, 1, tid, 256);
  fill_frags(Wfrag + 5120, Wc1, 18, 64, 64, 1, 4, tid, 256);
  fill_frags(Wfrag + 7168, Wc2, 64, 3,  3,  2, 1, tid, 256);
}

// ---------------------------------------------------------------------------
// Kernel 1: featurization. One thread per point.
// Writes X[n][64] f16: [0..31]=trilinear features, [32..34]=ray dir, rest 0.
// ---------------------------------------------------------------------------
__global__ void k_featurize(const float* __restrict__ rays_d,
                            const float* __restrict__ pts,
                            const float* __restrict__ times,
                            const float* __restrict__ grid_space,
                            const float* __restrict__ grid_time,
                            const float* __restrict__ features,
                            const int*   __restrict__ ridx,
                            _Float16*    __restrict__ X,
                            int N, long long Npad) {
  long long n = (long long)blockIdx.x * blockDim.x + threadIdx.x;
  if (n >= Npad) return;
  _Float16* xr = X + n * 64;
  if (n >= N) {                       // zero pad rows so WMMA tiles are clean
    for (int c = 0; c < 64; ++c) xr[c] = (_Float16)0.f;
    return;
  }

  // ---- time interpolation (12 channels) ----
  float t  = times[n];
  float xt = (t + 1.f) * 0.5f * (float)(TTT - 1);
  xt = fminf(fmaxf(xt, 0.f), (float)(TTT - 1));
  int ti0 = (int)floorf(xt);
  ti0 = min(max(ti0, 0), TTT - 2);
  float tw = xt - (float)ti0;
  float it[12];
#pragma unroll
  for (int j = 0; j < 12; ++j) {
    const float* g = grid_time + j * TTT;
    it[j] = g[ti0] * (1.f - tw) + g[ti0 + 1] * tw;
  }

  // ---- per-axis grid coords ----
  float p[3] = {pts[n * 3 + 0], pts[n * 3 + 1], pts[n * 3 + 2]};
  int   i0[3];
  float w[3];
#pragma unroll
  for (int k = 0; k < 3; ++k) {
    float x = (p[k] + 1.f) * 0.5f * (float)(RR - 1);
    x = fminf(fmaxf(x, 0.f), (float)(RR - 1));
    int i = (int)floorf(x);
    i = min(max(i, 0), RR - 2);
    i0[k] = i;
    w[k]  = x - (float)i;
  }

  // ---- tri-plane product (48 channels) ----
  const int ca[3] = {0, 0, 1};
  const int cb[3] = {1, 2, 2};
  float prod[48];
#pragma unroll
  for (int ch = 0; ch < 48; ++ch) prod[ch] = 1.f;
  for (int c = 0; c < 3; ++c) {
    int   ia = i0[ca[c]], ib = i0[cb[c]];
    float wa = w[ca[c]],  wb = w[cb[c]];
    float w00 = (1.f - wa) * (1.f - wb);
    float w01 = (1.f - wa) * wb;
    float w10 = wa * (1.f - wb);
    float w11 = wa * wb;
    const float* gp = grid_space + (size_t)c * 48 * RR * RR + (size_t)ia * RR + ib;
#pragma unroll 4
    for (int ch = 0; ch < 48; ++ch) {
      const float* gc = gp + (size_t)ch * RR * RR;
      float v = gc[0] * w00 + gc[1] * w01 + gc[RR] * w10 + gc[RR + 1] * w11;
      prod[ch] *= v;
    }
  }

  // ---- low-rank contraction -> 3D coords ----
  float coords[3];
#pragma unroll
  for (int od = 0; od < 3; ++od) {
    float acc = 0.f;
#pragma unroll
    for (int tr = 0; tr < 4; ++tr) {
      int base = od * 16 + tr * 4;
      float s = prod[base] + prod[base + 1] + prod[base + 2] + prod[base + 3];
      acc += s * it[od * 4 + tr];
    }
    coords[od] = fminf(fmaxf(acc, -1.f), 1.f);
  }

  // ---- trilinear feature sample (32 channels) ----
  int   fi[3];
  float fw[3];
#pragma unroll
  for (int k = 0; k < 3; ++k) {
    float x = (coords[k] + 1.f) * 0.5f * (float)(RR - 1);
    x = fminf(fmaxf(x, 0.f), (float)(RR - 1));
    int i = (int)floorf(x);
    i = min(max(i, 0), RR - 2);
    fi[k] = i;
    fw[k] = x - (float)i;
  }
  float wx0 = 1.f - fw[0], wx1 = fw[0];
  float wy0 = 1.f - fw[1], wy1 = fw[1];
  float wz0 = 1.f - fw[2], wz1 = fw[2];
  float w000 = wx0 * wy0 * wz0, w001 = wx0 * wy0 * wz1;
  float w010 = wx0 * wy1 * wz0, w011 = wx0 * wy1 * wz1;
  float w100 = wx1 * wy0 * wz0, w101 = wx1 * wy0 * wz1;
  float w110 = wx1 * wy1 * wz0, w111 = wx1 * wy1 * wz1;
  size_t base3 = ((size_t)fi[0] * RR + fi[1]) * RR + fi[2];
  const size_t SY = RR;
  const size_t SX = (size_t)RR * RR;
#pragma unroll 4
  for (int ch = 0; ch < FF; ++ch) {
    const float* fc = features + (size_t)ch * RR * RR * RR + base3;
    float v = fc[0]        * w000 + fc[1]          * w001 +
              fc[SY]       * w010 + fc[SY + 1]     * w011 +
              fc[SX]       * w100 + fc[SX + 1]     * w101 +
              fc[SX + SY]  * w110 + fc[SX + SY + 1]* w111;
    xr[ch] = (_Float16)v;
  }

  // ---- normalized ray direction ----
  int r = ridx[n];
  float dx = rays_d[r * 3 + 0], dy = rays_d[r * 3 + 1], dz = rays_d[r * 3 + 2];
  float inv = rsqrtf(dx * dx + dy * dy + dz * dz);
  xr[32] = (_Float16)(dx * inv);
  xr[33] = (_Float16)(dy * inv);
  xr[34] = (_Float16)(dz * inv);
#pragma unroll
  for (int c = 35; c < 64; ++c) xr[c] = (_Float16)0.f;
}

// ---------------------------------------------------------------------------
// Kernel 2: fused MLPs via WMMA. 256 threads = 8 waves; each wave processes
// TPW 16-point tiles. Weights arrive as a pre-fragmented 16KB f16 image that
// is copied into LDS with async global->LDS b128 ops when available.
// ---------------------------------------------------------------------------
#define TPW 4                 // tiles per wave per block
#define TPB (8 * TPW)         // tiles per block
#define ROWS_PER_BLOCK (TPB * 16)

__global__ void __launch_bounds__(256) k_mlp(
    const _Float16* __restrict__ X,
    const _Float16* __restrict__ Wfrag,
    const float* __restrict__ b1, const float* __restrict__ b2,
    const float* __restrict__ bc1, const float* __restrict__ bc2,
    float* __restrict__ sigmaB, float* __restrict__ rgbB) {
  __shared__ __align__(16) _Float16 wbuf[8192];      // fragment image (16KB)
  __shared__ __align__(16) _Float16 smh[8][16 * 64];
  __shared__ __align__(16) _Float16 smac[8][16 * 32];
  __shared__ __align__(16) float    sms[8][16 * 16];
  __shared__ float sb1[64], sb2[16], sbc1[64], sbc2[16];

  int tid = threadIdx.x;

#if HAVE_ASYNC_LDS
  {
    _Float16* gsrc = const_cast<_Float16*>(Wfrag);
#pragma unroll
    for (int r2 = 0; r2 < 4; ++r2) {
      int e = (r2 * 256 + tid) * 8;   // halves (16 bytes)
      __builtin_amdgcn_global_load_async_to_lds_b128(
          (v4i_t*)(gsrc + e), (v4i_t*)(wbuf + e), 0, 0);
    }
    __builtin_amdgcn_s_wait_asynccnt(0);
  }
#else
  {
#pragma unroll
    for (int r2 = 0; r2 < 4; ++r2) {
      int e = (r2 * 256 + tid) * 8;   // halves
      *reinterpret_cast<uint4*>(&wbuf[e]) =
          *reinterpret_cast<const uint4*>(&Wfrag[e]);
    }
  }
#endif
  for (int i = tid; i < 64; i += 256) { sb1[i] = b1[i]; sbc1[i] = bc1[i]; }
  if (tid < 16) { sb2[tid] = b2[tid]; sbc2[tid] = (tid < 3) ? bc2[tid] : 0.f; }
  __syncthreads();

  const _Float16* fW1  = wbuf + 0;
  const _Float16* fW2  = wbuf + 4096;
  const _Float16* fWc1 = wbuf + 5120;
  const _Float16* fWc2 = wbuf + 7168;

  int wave = tid >> 5, lane = tid & 31;
  int row = lane & 15, hi = lane >> 4;

  for (int itr = 0; itr < TPW; ++itr) {
    long long tile = (long long)blockIdx.x * TPB + itr * 8 + wave;
    const _Float16* Xg = X + tile * 16 * 64;
    __builtin_prefetch(Xg, 0, 1);

    // ---- layer 1: h = relu(X @ W1 + b1) : 16x64 ----
    v8f acc1[4];
#pragma unroll
    for (int nt = 0; nt < 4; ++nt) acc1[nt] = (v8f){0, 0, 0, 0, 0, 0, 0, 0};
#pragma unroll
    for (int kt = 0; kt < 2; ++kt) {
      v16h a = ld_afrag(Xg, row, 64, kt * 32, hi);
#pragma unroll
      for (int nt = 0; nt < 4; ++nt) {
        v16h b = ld_frag(fW1 + (kt * 4 + nt) * 512 + lane * 16);
        acc1[nt] = wmma16x16x32(a, b, acc1[nt]);
      }
    }
#pragma unroll
    for (int nt = 0; nt < 4; ++nt)
#pragma unroll
      for (int r = 0; r < 8; ++r) {
        int rowD = hi * 8 + r, colD = nt * 16 + row;
        float v = acc1[nt][r] + sb1[colD];
        smh[wave][rowD * 64 + colD] = (_Float16)(v > 0.f ? v : 0.f);
      }
    __syncthreads();

    // ---- layer 2: s = h @ W2 + b2 : 16x16 ----
    v8f sacc = (v8f){0, 0, 0, 0, 0, 0, 0, 0};
#pragma unroll
    for (int kt = 0; kt < 2; ++kt) {
      v16h a = ld_afrag(smh[wave], row, 64, kt * 32, hi);
      v16h b = ld_frag(fW2 + kt * 512 + lane * 16);
      sacc = wmma16x16x32(a, b, sacc);
    }
#pragma unroll
    for (int r = 0; r < 8; ++r)
      sms[wave][(hi * 8 + r) * 16 + row] = sacc[r] + sb2[row];
    __syncthreads();

    // sigma = exp(clip(s[:,0]))
    if (lane < 16) {
      float s0 = sms[wave][lane * 16];
      sigmaB[tile * 16 + lane] = expf(fminf(fmaxf(s0, -15.f), 15.f));
    }

    // ---- color input: [geo(15), rd(3), 0...] : 16x32 ----
    for (int idx = lane; idx < 512; idx += 32) {
      int r = idx >> 5, c = idx & 31;
      float v;
      if (c < 15)      v = sms[wave][r * 16 + c + 1];
      else if (c < 18) v = (float)Xg[r * 64 + 32 + (c - 15)];
      else             v = 0.f;
      smac[wave][idx] = (_Float16)v;
    }
    __syncthreads();

    // ---- color layer 1: hc = relu(ac @ Wc1 + bc1) : 16x64 ----
    v8f hacc[4];
#pragma unroll
    for (int nt = 0; nt < 4; ++nt) hacc[nt] = (v8f){0, 0, 0, 0, 0, 0, 0, 0};
    {
      v16h a = ld_afrag(smac[wave], row, 32, 0, hi);
#pragma unroll
      for (int nt = 0; nt < 4; ++nt) {
        v16h b = ld_frag(fWc1 + nt * 512 + lane * 16);
        hacc[nt] = wmma16x16x32(a, b, hacc[nt]);
      }
    }
    __syncthreads();              // done reading smh from layer 2
#pragma unroll
    for (int nt = 0; nt < 4; ++nt)
#pragma unroll
      for (int r = 0; r < 8; ++r) {
        int rowD = hi * 8 + r, colD = nt * 16 + row;
        float v = hacc[nt][r] + sbc1[colD];
        smh[wave][rowD * 64 + colD] = (_Float16)(v > 0.f ? v : 0.f);
      }
    __syncthreads();

    // ---- color layer 2: rgb = sigmoid(hc @ Wc2 + bc2) ----
    v8f cacc = (v8f){0, 0, 0, 0, 0, 0, 0, 0};
#pragma unroll
    for (int kt = 0; kt < 2; ++kt) {
      v16h a = ld_afrag(smh[wave], row, 64, kt * 32, hi);
      v16h b = ld_frag(fWc2 + kt * 512 + lane * 16);
      cacc = wmma16x16x32(a, b, cacc);
    }
    if (row < 3) {
#pragma unroll
      for (int r = 0; r < 8; ++r) {
        int rowD = hi * 8 + r;
        float v = cacc[r] + sbc2[row];
        rgbB[(tile * 16 + rowD) * 3 + row] = 1.f / (1.f + expf(-v));
      }
    }
    __syncthreads();              // protect smh/sms/smac before next iteration
  }
}

// ---------------------------------------------------------------------------
// Kernel 3: per-ray transmittance scan + composite. One block per ray.
// ---------------------------------------------------------------------------
__global__ void k_render(const float* __restrict__ sigmaB,
                         const float* __restrict__ rgbB,
                         const float* __restrict__ deltas,
                         const float* __restrict__ bg,
                         float* __restrict__ out, int S) {
  __shared__ float sc[128];
  __shared__ float red[4][128];
  int ray = blockIdx.x;
  int i = threadIdx.x;
  long long n = (long long)ray * S + i;

  float tau = sigmaB[n] * deltas[n];
  sc[i] = tau;
  __syncthreads();
  for (int off = 1; off < S; off <<= 1) {   // inclusive Hillis-Steele scan
    float v = (i >= off) ? sc[i - off] : 0.f;
    __syncthreads();
    sc[i] += v;
    __syncthreads();
  }
  float excl = sc[i] - tau;                  // exclusive within-ray cumsum
  float Tr = expf(-excl);
  float w = Tr * (1.f - expf(-tau));

  red[0][i] = w * rgbB[n * 3 + 0];
  red[1][i] = w * rgbB[n * 3 + 1];
  red[2][i] = w * rgbB[n * 3 + 2];
  red[3][i] = w;
  __syncthreads();
  for (int off = S >> 1; off > 0; off >>= 1) {
    if (i < off) {
      red[0][i] += red[0][i + off];
      red[1][i] += red[1][i + off];
      red[2][i] += red[2][i + off];
      red[3][i] += red[3][i + off];
    }
    __syncthreads();
  }
  if (i == 0) {
    float alpha = red[3][0];
#pragma unroll
    for (int c = 0; c < 3; ++c)
      out[ray * 3 + c] = red[c][0] + (1.f - alpha) * bg[ray * 3 + c];
  }
}

// ---------------------------------------------------------------------------
extern "C" void kernel_launch(void* const* d_in, const int* in_sizes, int n_in,
                              void* d_out, int out_size, void* d_ws, size_t ws_size,
                              hipStream_t stream) {
  const float* rays_d     = (const float*)d_in[0];
  const float* pts        = (const float*)d_in[1];
  const float* times      = (const float*)d_in[2];
  const float* deltas     = (const float*)d_in[3];
  const float* bg_color   = (const float*)d_in[4];
  const float* grid_space = (const float*)d_in[5];
  const float* grid_time  = (const float*)d_in[6];
  const float* features   = (const float*)d_in[7];
  const float* W1  = (const float*)d_in[8];
  const float* b1  = (const float*)d_in[9];
  const float* W2  = (const float*)d_in[10];
  const float* b2  = (const float*)d_in[11];
  const float* Wc1 = (const float*)d_in[12];
  const float* bc1 = (const float*)d_in[13];
  const float* Wc2 = (const float*)d_in[14];
  const float* bc2 = (const float*)d_in[15];
  const int*   ridx = (const int*)d_in[16];
  // d_in[17] (boundary) implied by fixed samples-per-ray layout.

  int n_rays = in_sizes[0] / 3;
  int N      = in_sizes[1] / 3;
  int S      = N / n_rays;                     // 128
  long long Npad = ((long long)N + ROWS_PER_BLOCK - 1) & ~(long long)(ROWS_PER_BLOCK - 1);

  char* ws = (char*)d_ws;
  _Float16* X = (_Float16*)ws;
  size_t off = ((size_t)Npad * 64 * sizeof(_Float16) + 255) & ~(size_t)255;
  float* sigmaB = (float*)(ws + off);
  off += ((size_t)Npad * sizeof(float) + 255) & ~(size_t)255;
  float* rgbB = (float*)(ws + off);
  off += ((size_t)Npad * 3 * sizeof(float) + 255) & ~(size_t)255;
  _Float16* Wfrag = (_Float16*)(ws + off);     // 8192 halves (16KB)

  k_prepw<<<1, 256, 0, stream>>>(W1, W2, Wc1, Wc2, Wfrag);

  int featThreads = 256;
  long long featBlocks = (Npad + featThreads - 1) / featThreads;
  k_featurize<<<(unsigned)featBlocks, featThreads, 0, stream>>>(
      rays_d, pts, times, grid_space, grid_time, features, ridx, X, N, Npad);

  long long mlpBlocks = Npad / ROWS_PER_BLOCK;
  k_mlp<<<(unsigned)mlpBlocks, 256, 0, stream>>>(
      X, Wfrag, b1, b2, bc1, bc2, sigmaB, rgbB);

  k_render<<<n_rays, S, 0, stream>>>(sigmaB, rgbB, deltas, bg_color,
                                     (float*)d_out, S);
}
